// MMMambaModel_27754078667293
// MI455X (gfx1250) — matrix operand order repaired and checked by reading, not compiled, not run on hardware
//
#include <hip/hip_runtime.h>
#include <hip/hip_bf16.h>

typedef __attribute__((ext_vector_type(16))) _Float16 v16h;
typedef __attribute__((ext_vector_type(8)))  _Float16 v8h;
typedef __attribute__((ext_vector_type(4)))  _Float16 v4h;
typedef __attribute__((ext_vector_type(8)))  float    v8f;

namespace {
constexpr int B_  = 128;
constexpr int S_  = 168;
constexpr int DM  = 128;
constexpr int I_  = 256;
constexpr int N_  = 16;
constexpr int R_  = 8;
constexpr int KC  = 4;            // conv taps
constexpr int P_  = 24;
constexpr int BS  = B_ * S_;      // 21504 rows; /16 = 1344 tiles
constexpr int MTI = BS / 16;      // 1344 row tiles
constexpr int SPLD = 48;          // padded leading dim for sp (R+2N = 40 -> 48)
}

// ---------------- WMMA fragment loads (CDNA5 wave32 layouts, f16 in memory) ---
// A 16x32: lane -> M = lane&15, half = lane>>4; kb = k0 + half*8.
// Fragment = halves [kb,kb+8) in VGPRs 0-3 and [kb+16,kb+24) in VGPRs 4-7:
// exactly two 16-byte loads, no cross-lane shuffling, no conversion.
__device__ inline v16h ld_a16(const _Float16* __restrict__ A, int lda, int m0, int k0,
                              int lane) {
  const int M  = m0 + (lane & 15);
  const int kb = k0 + ((lane >> 4) << 3);
  const _Float16* row = A + (long)M * lda + kb;
  const v8h lo = *(const v8h*)(row);
  const v8h hi = *(const v8h*)(row + 16);
  v16h a;
#pragma unroll
  for (int j = 0; j < 8; ++j) { a[j] = lo[j]; a[8 + j] = hi[j]; }
  return a;
}

// B 32x16 (B[k][n] = W[n][k], W row-major [Nout,Kdim] in f16):
// lane -> N = lane&15, half = lane>>4; kb = k0 + half*8, halves [kb,kb+8) and [kb+16,kb+24).
__device__ inline v16h ld_b16(const _Float16* __restrict__ W, int ldw, int n0, int k0,
                              int lane) {
  const int n  = n0 + (lane & 15);
  const int kb = k0 + ((lane >> 4) << 3);
  const _Float16* p = W + (long)n * ldw + kb;
  const v8h lo = *(const v8h*)(p);
  const v8h hi = *(const v8h*)(p + 16);
  v16h b;
#pragma unroll
  for (int j = 0; j < 8; ++j) { b[j] = lo[j]; b[8 + j] = hi[j]; }
  return b;
}

__device__ inline v8f wmma_f16(v16h a, v16h b, v8f c) {
  return __builtin_amdgcn_wmma_f32_16x16x32_f16(false, a, false, b, (short)0, c, false, false);
}

// ---------------- block reduction (wave32) ----------------
__device__ inline float blk_reduce_sum(float v, float* sm) {
#pragma unroll
  for (int o = 16; o > 0; o >>= 1) v += __shfl_xor(v, o, 32);
  const int nw = blockDim.x >> 5;
  const int w  = threadIdx.x >> 5;
  if ((threadIdx.x & 31) == 0) sm[w] = v;
  __syncthreads();
  float s = (threadIdx.x < nw) ? sm[threadIdx.x] : 0.f;
#pragma unroll
  for (int o = 16; o > 0; o >>= 1) s += __shfl_xor(s, o, 32);
  if (threadIdx.x == 0) sm[0] = s;
  __syncthreads();
  const float r = sm[0];
  __syncthreads();
  return r;
}

// ---------------- one-shot f32 -> f16 conversion (vector x4) ----------------
__global__ __launch_bounds__(256) void mm_cvt16_k(const float* __restrict__ src,
                                                  _Float16* __restrict__ dst, long n4) {
  const long i = (long)blockIdx.x * blockDim.x + threadIdx.x;
  if (i >= n4) return;
  const float4 v = ((const float4*)src)[i];
  v4h o;
  o[0] = (_Float16)v.x; o[1] = (_Float16)v.y;
  o[2] = (_Float16)v.z; o[3] = (_Float16)v.w;
  ((v4h*)dst)[i] = o;
}

// pad + convert xp_w [40,256] -> f16 [48,256] (zero tail rows)
__global__ __launch_bounds__(256) void mm_padw_k(const float* __restrict__ src,
                                                 _Float16* __restrict__ dst) {
  const int idx = blockIdx.x * blockDim.x + threadIdx.x;
  if (idx >= SPLD * I_) return;
  const int n = idx / I_;
  dst[idx] = (n < R_ + 2 * N_) ? (_Float16)src[n * I_ + (idx % I_)] : (_Float16)0.f;
}

// ---------------- templated GEMM: D[BS,NOUT] = A16 @ W16.T (+bias, +=) -------
// One wave computes a 16 x (16*NTPW) output block; A fragment reused NTPW times.
template <int KDIM, int NOUT, int NTPW, bool RES>
__global__ __launch_bounds__(256) void mm_gemm_t(const _Float16* __restrict__ A,
                                                 const _Float16* __restrict__ W,
                                                 const float* __restrict__ bias,
                                                 float* __restrict__ D) {
  constexpr int NWG = NOUT / (16 * NTPW);
  const int wid  = blockIdx.x * 8 + (threadIdx.x >> 5);
  const int lane = threadIdx.x & 31;
  const int mt = wid / NWG, ng = wid % NWG;
  if (mt >= MTI) return;
  const int m0 = mt * 16, n0 = ng * (16 * NTPW);
  const _Float16* arow = A + (long)(m0 + (lane & 15)) * KDIM;
  v8f zero = {};
  v8f c[NTPW];
#pragma unroll
  for (int j = 0; j < NTPW; ++j) c[j] = zero;
  for (int k0 = 0; k0 < KDIM; k0 += 32) {
    __builtin_prefetch(arow + k0 + 32, 0, 0);   // speculative; OOB dropped by HW
    v16h a = ld_a16(A, KDIM, m0, k0, lane);
#pragma unroll
    for (int j = 0; j < NTPW; ++j) {
      v16h b = ld_b16(W, KDIM, n0 + 16 * j, k0, lane);
      c[j] = wmma_f16(a, b, c[j]);
    }
  }
  const int half = lane >> 4;
#pragma unroll
  for (int j = 0; j < NTPW; ++j) {
    const int n = n0 + 16 * j + (lane & 15);
    const float bv = bias ? bias[n] : 0.f;
#pragma unroll
    for (int r = 0; r < 8; ++r) {
      const int  M   = m0 + r + 8 * half;
      const long idx = (long)M * NOUT + n;
      if (RES) D[idx] = D[idx] + c[j][r] + bv;
      else     D[idx] = c[j][r] + bv;
    }
  }
}

// ---------------- kernel: feature fusion ----------------
// h[BS,DM] = x_load@Wl.T + x_img@Wi.T + x_text@Wt.T + b_fuse   (4 N-tiles/wave)
__global__ __launch_bounds__(256) void mm_fuse_k(const float* __restrict__ xl,
                                                 const _Float16* __restrict__ xi,
                                                 const _Float16* __restrict__ xt,
                                                 const float* __restrict__ Wl,
                                                 const _Float16* __restrict__ Wi,
                                                 const _Float16* __restrict__ Wt,
                                                 const float* __restrict__ bf,
                                                 float* __restrict__ h) {
  constexpr int NTPW = 4;
  constexpr int NWG  = DM / (16 * NTPW);     // 2
  const int wid  = blockIdx.x * 8 + (threadIdx.x >> 5);
  const int lane = threadIdx.x & 31;
  const int mt = wid / NWG, ng = wid % NWG;
  if (mt >= MTI) return;
  const int m0 = mt * 16, n0 = ng * (16 * NTPW);
  const _Float16* trow = xt + (long)(m0 + (lane & 15)) * 768;
  v8f zero = {};
  v8f c[NTPW] = {zero, zero, zero, zero};
  for (int k0 = 0; k0 < 768; k0 += 32) {
    __builtin_prefetch(trow + k0 + 32, 0, 0);
    v16h a = ld_a16(xt, 768, m0, k0, lane);
#pragma unroll
    for (int j = 0; j < NTPW; ++j)
      c[j] = wmma_f16(a, ld_b16(Wt, 768, n0 + 16 * j, k0, lane), c[j]);
  }
#pragma unroll
  for (int k0 = 0; k0 < 64; k0 += 32) {
    v16h a = ld_a16(xi, 64, m0, k0, lane);
#pragma unroll
    for (int j = 0; j < NTPW; ++j)
      c[j] = wmma_f16(a, ld_b16(Wi, 64, n0 + 16 * j, k0, lane), c[j]);
  }
  const int half = lane >> 4;
#pragma unroll
  for (int j = 0; j < NTPW; ++j) {
    const int n = n0 + 16 * j + (lane & 15);
    const float wl = Wl[n];                 // Wl is [DM,1]
    const float bv = bf[n];
#pragma unroll
    for (int r = 0; r < 8; ++r) {
      const int M = m0 + r + 8 * half;
      h[(long)M * DM + n] = c[j][r] + xl[M] * wl + bv;
    }
  }
}

// ---------------- kernel: rmsnorm (per token row) -> f16 for in-proj --------
__global__ __launch_bounds__(128) void mm_rmsnorm_k(const float* __restrict__ h,
                                                    const float* __restrict__ w,
                                                    _Float16* __restrict__ out) {
  __shared__ float sm[4];
  const long row = blockIdx.x;
  const int  t   = threadIdx.x;
  const float x  = h[row * DM + t];
  const float ms = blk_reduce_sum(x * x, sm) / DM;
  out[row * DM + t] = (_Float16)(x * rsqrtf(ms + 1e-5f) * w[t]);
}

// ---------------- kernel: causal depthwise conv + SiLU (f32 + f16 outputs) ---
__global__ __launch_bounds__(256) void mm_conv_silu_k(const float* __restrict__ hs,
                                                      const float* __restrict__ cw,
                                                      const float* __restrict__ cb,
                                                      float* __restrict__ ut,
                                                      _Float16* __restrict__ ut16) {
  const long idx = (long)blockIdx.x * blockDim.x + threadIdx.x;
  if (idx >= (long)BS * I_) return;
  const int  i  = (int)(idx & (I_ - 1));
  const long bs = idx >> 8;
  const int  s  = (int)(bs % S_);
  const long b  = bs / S_;
  float acc = cb[i];
#pragma unroll
  for (int k = 0; k < KC; ++k) {
    const int sk = s + k - (KC - 1);
    if (sk >= 0) acc += cw[i * KC + k] * hs[(b * S_ + sk) * (2 * I_) + i];
  }
  const float sig = 1.f / (1.f + __expf(-acc));
  const float u   = acc * sig;
  ut[idx]   = u;                 // f32 for the scan (precision)
  ut16[idx] = (_Float16)u;       // f16 for the x-proj GEMM
}

// ---------------- kernel: fused dt/softplus + SSM scan + gate (f32) ----------
__global__ __launch_bounds__(256) void mm_scan_k(const float* __restrict__ sp,
                                                 const float* __restrict__ ut,
                                                 const float* __restrict__ hsg,
                                                 const float* __restrict__ dtw,
                                                 const float* __restrict__ dtb,
                                                 const float* __restrict__ alog,
                                                 const float* __restrict__ Dp,
                                                 _Float16* __restrict__ y16) {
  const int b = blockIdx.x;
  const int i = threadIdx.x;
  float dw[R_];
#pragma unroll
  for (int r = 0; r < R_; ++r) dw[r] = dtw[i * R_ + r];
  const float db = dtb[i];
  float Ai[N_];
#pragma unroll
  for (int n = 0; n < N_; ++n) Ai[n] = -__expf(alog[i * N_ + n]);
  const float dpi = Dp[i];
  float st[N_];
#pragma unroll
  for (int n = 0; n < N_; ++n) st[n] = 0.f;

  __shared__ float row[SPLD];
  for (int s = 0; s < S_; ++s) {
    const long bs = (long)b * S_ + s;
    if (threadIdx.x < SPLD) row[threadIdx.x] = sp[bs * SPLD + threadIdx.x];
    __syncthreads();
    float din = db;
#pragma unroll
    for (int r = 0; r < R_; ++r) din += dw[r] * row[r];
    const float dt  = (din > 20.f) ? din : __logf(1.f + __expf(din));   // softplus
    const float u   = ut[bs * I_ + i];
    const float dtu = dt * u;
    float ys = 0.f;
#pragma unroll
    for (int n = 0; n < N_; ++n) {
      const float dA = __expf(dt * Ai[n]);
      st[n] = dA * st[n] + dtu * row[R_ + n];          // B from sp[8..23]
      ys += st[n] * row[R_ + N_ + n];                  // C from sp[24..39]
    }
    ys += u * dpi;
    const float g  = hsg[bs * (2 * I_) + I_ + i];      // gate = cols [I,2I)
    const float sg = g / (1.f + __expf(-g));
    y16[bs * I_ + i] = (_Float16)(ys * sg);            // f16 for out-proj GEMM
    __syncthreads();
  }
}

// ---------------- kernel: final rmsnorm -> layernorm -> FC on last token -----
__global__ __launch_bounds__(128) void mm_head_k(const float* __restrict__ h,
                                                 const float* __restrict__ fw,
                                                 const float* __restrict__ lw,
                                                 const float* __restrict__ lb,
                                                 const float* __restrict__ fcw,
                                                 const float* __restrict__ fcb,
                                                 float* __restrict__ out) {
  __shared__ float sm[4];
  __shared__ float buf[DM];
  const int  b   = blockIdx.x;
  const int  t   = threadIdx.x;
  const long row = (long)b * S_ + (S_ - 1);
  const float x  = h[row * DM + t];
  const float ms = blk_reduce_sum(x * x, sm) / DM;
  const float xr = x * rsqrtf(ms + 1e-5f) * fw[t];
  const float mu = blk_reduce_sum(xr, sm) / DM;
  const float d  = xr - mu;
  const float va = blk_reduce_sum(d * d, sm) / DM;
  buf[t] = d * rsqrtf(va + 1e-5f) * lw[t] + lb[t];
  __syncthreads();
  if (t < P_) {
    float acc = fcb[t];
#pragma unroll 4
    for (int k = 0; k < DM; ++k) acc += fcw[t * DM + k] * buf[k];
    out[b * P_ + t] = acc;
  }
}

// ---------------- host-side launch ----------------
extern "C" void kernel_launch(void* const* d_in, const int* in_sizes, int n_in,
                              void* d_out, int out_size, void* d_ws, size_t ws_size,
                              hipStream_t stream) {
  (void)in_sizes; (void)n_in; (void)out_size; (void)ws_size;
  const float* xl    = (const float*)d_in[0];
  const float* xi    = (const float*)d_in[1];
  const float* xt    = (const float*)d_in[2];
  const float* Wl    = (const float*)d_in[3];
  const float* Wi    = (const float*)d_in[4];
  const float* Wt    = (const float*)d_in[5];
  const float* bf    = (const float*)d_in[6];
  const float* mnw   = (const float*)d_in[7];
  const float* inw   = (const float*)d_in[8];
  const float* inb   = (const float*)d_in[9];
  const float* cw    = (const float*)d_in[10];
  const float* cb    = (const float*)d_in[11];
  const float* xpw   = (const float*)d_in[12];
  const float* dtw   = (const float*)d_in[13];
  const float* dtb   = (const float*)d_in[14];
  const float* alog  = (const float*)d_in[15];
  const float* Dp    = (const float*)d_in[16];
  const float* outw  = (const float*)d_in[17];
  const float* outb  = (const float*)d_in[18];
  const float* fnw   = (const float*)d_in[19];
  const float* lnw   = (const float*)d_in[20];
  const float* lnb   = (const float*)d_in[21];
  const float* fcw   = (const float*)d_in[22];
  const float* fcb   = (const float*)d_in[23];
  float* out = (float*)d_out;

  char* ws = (char*)d_ws;
  size_t off = 0;
  auto alloc = [&](size_t bytes) { void* p = (void*)(ws + off);
                                   off += (bytes + 255) & ~size_t(255); return p; };
  // f32 buffers
  float* h   = (float*)alloc((size_t)BS * DM * 4);       // residual stream
  float* hsg = (float*)alloc((size_t)BS * 2 * I_ * 4);   // in-proj out [hs|gate]
  float* ut  = (float*)alloc((size_t)BS * I_ * 4);       // conv+silu (f32, scan)
  float* sp  = (float*)alloc((size_t)BS * SPLD * 4);     // x-proj out (padded 48)
  // f16 GEMM operands
  _Float16* xt16  = (_Float16*)alloc((size_t)BS * 768 * 2);
  _Float16* xi16  = (_Float16*)alloc((size_t)BS * 64 * 2);
  _Float16* Wt16  = (_Float16*)alloc((size_t)DM * 768 * 2);
  _Float16* Wi16  = (_Float16*)alloc((size_t)DM * 64 * 2);
  _Float16* hn16  = (_Float16*)alloc((size_t)BS * DM * 2);
  _Float16* ut16  = (_Float16*)alloc((size_t)BS * I_ * 2);
  _Float16* yb16  = (_Float16*)alloc((size_t)BS * I_ * 2);
  _Float16* inw16 = (_Float16*)alloc((size_t)2 * I_ * DM * 2);
  _Float16* xpp16 = (_Float16*)alloc((size_t)SPLD * I_ * 2);
  _Float16* ow16  = (_Float16*)alloc((size_t)DM * I_ * 2);

  auto cvt = [&](const float* s, _Float16* d, long n) {
    const long n4 = n / 4;
    mm_cvt16_k<<<(unsigned)((n4 + 255) / 256), 256, 0, stream>>>(s, d, n4);
  };

  // one-shot conversions of fusion operands
  cvt(xt, xt16, (long)BS * 768);
  cvt(xi, xi16, (long)BS * 64);
  cvt(Wt, Wt16, (long)DM * 768);
  cvt(Wi, Wi16, (long)DM * 64);

  // fusion: 1344 M-tiles x 2 wave-groups (4 N-tiles/wave), 8 waves/block
  mm_fuse_k<<<MTI * 2 / 8, 256, 0, stream>>>(xl, xi16, xt16, Wl, Wi16, Wt16, bf, h);

  for (int l = 0; l < 2; ++l) {
    cvt(inw + (size_t)l * 2 * I_ * DM, inw16, (long)2 * I_ * DM);
    cvt(outw + (size_t)l * DM * I_, ow16, (long)DM * I_);
    mm_padw_k<<<(SPLD * I_ + 255) / 256, 256, 0, stream>>>(
        xpw + (size_t)l * (R_ + 2 * N_) * I_, xpp16);

    mm_rmsnorm_k<<<BS, DM, 0, stream>>>(h, mnw + l * DM, hn16);
    // in-proj: [BS,128] -> [BS,512], 4 N-tiles/wave -> 8 groups, 1344 blocks
    mm_gemm_t<DM, 2 * I_, 4, false><<<MTI * 8 / 8, 256, 0, stream>>>(
        hn16, inw16, inb + (size_t)l * 2 * I_, hsg);
    mm_conv_silu_k<<<((long)BS * I_ + 255) / 256, 256, 0, stream>>>(
        hsg, cw + (size_t)l * I_ * KC, cb + (size_t)l * I_, ut, ut16);
    // x-proj: unguarded 3 N-tiles = full padded N (168 blocks)
    mm_gemm_t<I_, SPLD, 3, false><<<MTI / 8, 256, 0, stream>>>(ut16, xpp16, nullptr, sp);
    mm_scan_k<<<B_, I_, 0, stream>>>(
        sp, ut, hsg, dtw + (size_t)l * I_ * R_, dtb + (size_t)l * I_,
        alog + (size_t)l * I_ * N_, Dp + (size_t)l * I_, yb16);
    // out-proj + residual: h += y @ out_w.T + out_b (4 N-tiles/wave, 336 blocks)
    mm_gemm_t<I_, DM, 4, true><<<MTI * 2 / 8, 256, 0, stream>>>(
        yb16, ow16, outb + (size_t)l * DM, h);
  }

  mm_head_k<<<B_, DM, 0, stream>>>(h, fnw, lnw, lnb, fcw, fcb, out);
}
